// SimpleGC_32134945308883
// MI455X (gfx1250) — matrix operand order repaired and checked by reading, not compiled, not run on hardware
//
#include <hip/hip_runtime.h>

// Problem constants (from reference setup_inputs)
#define N_  32
#define C_  64
#define V_  25
#define T_  1024
#define K_  3
#define M_  (V_ * C_)   // 1600  (rows of Mfull = (v,c))
#define KD_ (V_ * C_)   // 1600  (cols of Mfull = (w,c'))

// GEMM tiling: block = 8 wave32 arranged 2(M) x 4(N); wave tile 32x64
#define BM 64
#define BN 256
#define BK 32
#define BSTRIDE 40      // padded k-stride (elems) for B tile in LDS
#define ASZ (BM * BK)        // one A buffer (elems)
#define BSZ (BN * BSTRIDE)   // one B buffer (elems)

typedef __attribute__((ext_vector_type(16))) __bf16 v16bf;
typedef __attribute__((ext_vector_type(8)))  __bf16 v8bf;
typedef __attribute__((ext_vector_type(2)))  __bf16 v2bf;
typedef __attribute__((ext_vector_type(8)))  float  v8f;
typedef __attribute__((ext_vector_type(4)))  float  v4f;

__device__ __forceinline__ unsigned pack_bf16x2(float a, float b) {
  v2bf p;
  p[0] = (__bf16)a;
  p[1] = (__bf16)b;          // fuses to v_cvt_pk_bf16_f32
  return __builtin_bit_cast(unsigned, p);
}

// ---------------------------------------------------------------------------
// Kernel 1: fold adjacency A into weights W.
// Mfull[(v*C+c), (w*C+c2)] = sum_k A[k,w,v] * W[w, c*K+k, c2]   (stored bf16)
// ---------------------------------------------------------------------------
__global__ __launch_bounds__(256)
void build_mfull(const float* __restrict__ W, const float* __restrict__ A,
                 __bf16* __restrict__ Mfull) {
  int g = blockIdx.x * blockDim.x + threadIdx.x;
  if (g >= M_ * KD_) return;
  int m = g / KD_;
  int r = g - m * KD_;
  int v  = m >> 6;
  int c  = m & 63;
  int w  = r >> 6;
  int c2 = r & 63;
  float acc = 0.f;
#pragma unroll
  for (int k = 0; k < K_; ++k) {
    float a  = A[(k * V_ + w) * V_ + v];
    float wt = W[w * (K_ * C_ * C_) + (c * K_ + k) * C_ + c2];
    acc += a * wt;
  }
  Mfull[g] = (__bf16)acc;
}

// ---------------------------------------------------------------------------
// Kernel 2: batched GEMM  out[n] (1600 x 1024) = Mfull (1600x1600) @ X[n]
// X[n][(w*C+c2), t] = x[n, c2, w, t]   (fp32 in global, bf16 in LDS)
// out row m = v*C+c  ->  out[n, c, v, t]
// Double-buffered LDS (ping-pong) + register pipeline: one barrier per
// K-step; globals for tile k+1 load while tile k is multiplied.
// ---------------------------------------------------------------------------
__global__ __launch_bounds__(256)
void gemm_gc(const float* __restrict__ x, const __bf16* __restrict__ Mfull,
             float* __restrict__ out) {
  __shared__ __align__(16) __bf16 ldsA[2 * ASZ];  // [buf][row][k], stride 32
  __shared__ __align__(16) __bf16 ldsB[2 * BSZ];  // [buf][col][k], stride 40

  const int n  = blockIdx.z;
  const int m0 = blockIdx.y * BM;
  const int t0 = blockIdx.x * BN;

  const int tid   = threadIdx.x;
  const int lane  = tid & 31;
  const int wave  = tid >> 5;      // 0..7
  const int waveM = wave & 1;      // 2 M-groups of 32 rows
  const int waveN = wave >> 1;     // 4 N-groups of 64 cols
  const int half  = lane >> 4;     // 0/1
  const int l16   = lane & 15;

  const float* xn = x + (size_t)n * (C_ * V_ * T_);

  // Per-thread staging coordinates (loop invariant)
  const int aRow   = (tid * 8) >> 5;       // A tile row this thread stages
  const int aKK    = (tid * 8) & (BK - 1); // A tile k offset
  const int tqB    = tid & 63;             // B t-quad
  const int rqBase = tid >> 6;             // B row-quad base (0..3)
  const int tB     = t0 + tqB * 4;

  // Register-resident staging buffers for the *next* K-tile
  v8bf aReg;
  v4f  fReg[2][4];                         // [it][k-sub] 4 t-values each

  auto load_tile = [&](int k0n) {
    aReg = *(const v8bf*)(Mfull + (size_t)(m0 + aRow) * KD_ + (k0n + aKK));
#pragma unroll
    for (int it = 0; it < 2; ++it) {
      int kk = (rqBase + it * 4) * 4;
#pragma unroll
      for (int j = 0; j < 4; ++j) {
        int r  = k0n + kk + j;
        int w  = r >> 6;
        int c2 = r & 63;
        fReg[it][j] =
            *(const v4f*)(xn + (size_t)c2 * (V_ * T_) + (size_t)w * T_ + tB);
      }
    }
    // warm L2 for the A stream one tile further ahead
    if (k0n + BK < KD_)
      __builtin_prefetch(Mfull + (size_t)(m0 + aRow) * KD_ + (k0n + BK + aKK),
                         0, 1);
  };

  auto store_tile = [&](int buf) {
    *(v8bf*)(ldsA + buf * ASZ + aRow * BK + aKK) = aReg;
    __bf16* bb = ldsB + buf * BSZ;
#pragma unroll
    for (int it = 0; it < 2; ++it) {
      int kk = (rqBase + it * 4) * 4;
#pragma unroll
      for (int i = 0; i < 4; ++i) {
        uint2 d;
        d.x = pack_bf16x2(fReg[it][0][i], fReg[it][1][i]);  // k = kk, kk+1
        d.y = pack_bf16x2(fReg[it][2][i], fReg[it][3][i]);  // k = kk+2, kk+3
        *(uint2*)(bb + (size_t)(tqB * 4 + i) * BSTRIDE + kk) = d;
      }
    }
  };

  v8f acc[2][4];
#pragma unroll
  for (int sm = 0; sm < 2; ++sm)
#pragma unroll
    for (int sn = 0; sn < 4; ++sn)
#pragma unroll
      for (int i = 0; i < 8; ++i)
        acc[sm][sn][i] = 0.f;

  // Prologue: stage tile 0 into buffer 0
  load_tile(0);
  store_tile(0);
  __syncthreads();

  int p = 0;
  for (int k0 = 0; k0 < KD_; k0 += BK, p ^= 1) {
    const bool has_next = (k0 + BK) < KD_;
    // Issue global loads for tile k+1 early (latency hidden behind WMMA)
    if (has_next) load_tile(k0 + BK);

    const __bf16* la = ldsA + p * ASZ;
    const __bf16* lb = ldsB + p * BSZ;

    // --- A fragments: lane row = l16; K = 16*half + j (contiguous per row)
    v16bf afrag[2];
#pragma unroll
    for (int sm = 0; sm < 2; ++sm) {
      int row = waveM * 32 + sm * 16 + l16;
      v8bf lo = *(const v8bf*)(la + row * BK + half * 16);
      v8bf hi = *(const v8bf*)(la + row * BK + half * 16 + 8);
      afrag[sm] = __builtin_shufflevector(lo, hi, 0, 1, 2, 3, 4, 5, 6, 7,
                                          8, 9, 10, 11, 12, 13, 14, 15);
    }
    // --- B fragments: lane col = l16; K = 16*half + j (contiguous per col)
    v16bf bfrag[4];
#pragma unroll
    for (int sn = 0; sn < 4; ++sn) {
      int col = waveN * 64 + sn * 16 + l16;
      const __bf16* base = lb + (size_t)col * BSTRIDE + half * 16;
      v8bf lo = *(const v8bf*)(base);
      v8bf hi = *(const v8bf*)(base + 8);
      bfrag[sn] = __builtin_shufflevector(lo, hi, 0, 1, 2, 3, 4, 5, 6, 7,
                                          8, 9, 10, 11, 12, 13, 14, 15);
    }

#pragma unroll
    for (int sm = 0; sm < 2; ++sm)
#pragma unroll
      for (int sn = 0; sn < 4; ++sn)
        acc[sm][sn] = __builtin_amdgcn_wmma_f32_16x16x32_bf16(
            /*neg_a=*/false, afrag[sm], /*neg_b=*/false, bfrag[sn],
            /*c_mod=*/(short)0, acc[sm][sn], /*reuse_a=*/false, /*reuse_b=*/false);

    // Write tile k+1 into the other buffer; single barrier per iteration.
    if (has_next) store_tile(p ^ 1);
    __syncthreads();
  }

  // --- write back (non-temporal: out is write-once, keep L2 for x/Mfull)
#pragma unroll
  for (int sm = 0; sm < 2; ++sm)
#pragma unroll
    for (int sn = 0; sn < 4; ++sn)
#pragma unroll
      for (int i = 0; i < 8; ++i) {
        int m = m0 + waveM * 32 + sm * 16 + half * 8 + i;
        int t = t0 + waveN * 64 + sn * 16 + l16;
        int v = m >> 6;
        int c = m & 63;
        __builtin_nontemporal_store(
            acc[sm][sn][i],
            out + (size_t)n * (C_ * V_ * T_) + (size_t)c * (V_ * T_) +
                (size_t)v * T_ + t);
      }
}

// ---------------------------------------------------------------------------
extern "C" void kernel_launch(void* const* d_in, const int* in_sizes, int n_in,
                              void* d_out, int out_size, void* d_ws, size_t ws_size,
                              hipStream_t stream) {
  const float* x = (const float*)d_in[0];   // [N, C, V, T]
  const float* W = (const float*)d_in[1];   // [V, K*C, C]
  const float* A = (const float*)d_in[2];   // [K, V, V]
  float* out = (float*)d_out;               // [N, C, V, T]
  __bf16* Mfull = (__bf16*)d_ws;            // 1600*1600 bf16 = 5.12 MB

  int total = M_ * KD_;
  build_mfull<<<(total + 255) / 256, 256, 0, stream>>>(W, A, Mfull);

  dim3 grid(T_ / BN, M_ / BM, N_);          // (4, 25, 32)
  gemm_gc<<<grid, 256, 0, stream>>>(x, Mfull, out);
}
